// PseudoRegularization_1331439862526
// MI455X (gfx1250) — matrix operand order repaired and checked by reading, not compiled, not run on hardware
//
#include <hip/hip_runtime.h>

// MI455X / gfx1250 implementation.
// Pipeline (all on `stream`):
//   k_prep_*   : NCHW fp32 -> NHWC bf16 activation staging + weight packing
//   k_convT    : ConvTranspose2d(32,32,4,2,1) as per-parity GEMM, 4 x wmma_bf16 K-steps
//   k_conv7x1  : 7x1 conv 32->49(pad 64); block stages 7x32x32 bf16 strip in LDS
//                (global_load_async_to_lds_b128, ASYNCcnt-tracked), 7 x wmma K-steps/wave
//   k_conv1x7  : 1x7 conv 49->49(pad 64); block stages 22x64 bf16 strip in LDS,
//                14 x wmma K-steps/wave, fused softmax + 7x7 flow gather -> [B,2,H,W] fp32
// Workspace (~208 MB): rf 16MB | f0 64MB | f1 128MB | packed weights ~92KB.

typedef unsigned short u16;
typedef __attribute__((ext_vector_type(8)))  u16    v8us;
typedef __attribute__((ext_vector_type(16))) u16    v16us;
typedef __attribute__((ext_vector_type(16))) __bf16 v16bf;
typedef __attribute__((ext_vector_type(8)))  float  v8f;
typedef int gvi4 __attribute__((vector_size(16)));   // matches async builtin param type

#define B_N   4
#define CH32  32
#define HH    256
#define WH    256
#define HF    512
#define WF    512
#define C49   49
#define C64   64

#define OFF_RF  (size_t)0
#define OFF_F0  (size_t)16777216
#define OFF_F1  (size_t)83886080
#define OFF_WTB (size_t)218103808
#define OFF_W1B (OFF_WTB + 8192)
#define OFF_W2B (OFF_W1B + 28672)

#if defined(__AMDGCN__) && __has_builtin(__builtin_amdgcn_global_load_async_to_lds_b128)
#define HAS_ASYNC 1
#else
#define HAS_ASYNC 0
#endif

#define AS1 __attribute__((address_space(1)))
#define AS3 __attribute__((address_space(3)))

__device__ __forceinline__ u16 f2bf(float f) {
  unsigned u = __float_as_uint(f);
  u += 0x7FFFu + ((u >> 16) & 1u);          // round-to-nearest-even
  return (u16)(u >> 16);
}

__device__ __forceinline__ v8us zero8() {
  v8us z;
#pragma unroll
  for (int i = 0; i < 8; ++i) z[i] = 0;
  return z;
}

__device__ __forceinline__ v16bf mk_frag(v8us lo, v8us hi) {
  v16us c = __builtin_shufflevector(lo, hi, 0,1,2,3,4,5,6,7,8,9,10,11,12,13,14,15);
  return __builtin_bit_cast(v16bf, c);
}

__device__ __forceinline__ v8f wmma_bf16(v16bf a, v16bf b, v8f c) {
  return __builtin_amdgcn_wmma_f32_16x16x32_bf16(false, a, false, b, (short)0, c, false, false);
}

// 16-byte global -> LDS copy via the CDNA5 async-to-LDS path (ASYNCcnt-tracked).
__device__ __forceinline__ void cp16_g2l(u16* l, const u16* g) {
#if HAS_ASYNC
  __builtin_amdgcn_global_load_async_to_lds_b128(
      (AS1 gvi4*)(unsigned long long)g,
      (AS3 gvi4*)(unsigned int)(unsigned long long)l,
      0, 0);
#else
  *(v8us*)l = *(const v8us*)g;
#endif
}

__device__ __forceinline__ void async_fence() {
#if HAS_ASYNC
  asm volatile("s_wait_asynccnt 0x0" ::: "memory");
#endif
}

// ---------------- prep kernels ----------------

// reg_feat [4][32][256][256] fp32 -> rf NHWC bf16 [4][256][256][32]
__global__ void k_prep_rf(const float* __restrict__ src, u16* __restrict__ rf) {
  int idx = blockIdx.x * blockDim.x + threadIdx.x;   // 8388608
  int j = idx & 255, i = (idx >> 8) & 255, c = (idx >> 16) & 31, b = idx >> 21;
  rf[(((b * HH + i) * WH + j) << 5) + c] = f2bf(src[idx]);
}

// wt [ci=32][co=32][kh=4][kw=4] -> wtb [kh][kw][co][ci]
__global__ void k_prep_wt(const float* __restrict__ wt, u16* __restrict__ wtb) {
  int idx = blockIdx.x * blockDim.x + threadIdx.x;   // 16384
  int ci = idx & 31, co = (idx >> 5) & 31, kw = (idx >> 10) & 3, kh = idx >> 12;
  wtb[idx] = f2bf(wt[((ci * 32 + co) * 4 + kh) * 4 + kw]);
}

// w1 [o=49][c=32][r=7][1] -> w1b [r][o64][c32] (o>=49 -> 0)
__global__ void k_prep_w1(const float* __restrict__ w1, u16* __restrict__ w1b) {
  int idx = blockIdx.x * blockDim.x + threadIdx.x;   // 14336
  int c = idx & 31, o = (idx >> 5) & 63, r = idx >> 11;
  float v = (o < C49) ? w1[(o * 32 + c) * 7 + r] : 0.f;
  w1b[idx] = f2bf(v);
}

// w2 [o=49][cc=49][1][s=7] -> w2b [s][chalf][o64][c32] (pad -> 0)
__global__ void k_prep_w2(const float* __restrict__ w2, u16* __restrict__ w2b) {
  int idx = blockIdx.x * blockDim.x + threadIdx.x;   // 28672
  int c = idx & 31, o = (idx >> 5) & 63, sc = idx >> 11;
  int s = sc >> 1, ch = sc & 1, cc = ch * 32 + c;
  float v = (o < C49 && cc < C49) ? w2[(o * C49 + cc) * 7 + s] : 0.f;
  w2b[idx] = f2bf(v);
}

// ---------------- stage 1: ConvTranspose2d(32,32,4,s2,p1) ----------------
// One wave per (b, y, px, xm-tile, ntile). M=16 same-parity pixels, N=16 co, K=128.
// rf is 16 MB -> fully L2-resident; per-lane loads hit L2.
__global__ void k_convT(const u16* __restrict__ rf, const u16* __restrict__ wtb,
                        const float* __restrict__ bt, u16* __restrict__ f0) {
  int wave = blockIdx.x * (blockDim.x >> 5) + (threadIdx.x >> 5);
  int lane = threadIdx.x & 31;
  int ntile = wave & 1;  wave >>= 1;
  int tj    = wave & 15; wave >>= 4;
  int px    = wave & 1;  wave >>= 1;
  int y     = wave & 511; wave >>= 9;
  int b     = wave;
  int m = lane & 15, khalf = lane >> 4;
  int kh0 = (y + 1) & 1, kw0 = (px + 1) & 1;
  int xm0 = tj * 16;

  v8f acc;
#pragma unroll
  for (int g = 0; g < 8; ++g) acc[g] = 0.f;

#pragma unroll
  for (int t = 0; t < 4; ++t) {
    int kh = kh0 + 2 * (t >> 1);
    int kw = kw0 + 2 * (t & 1);
    int i  = (y + 1 - kh) / 2;            // exact (parity match)
    int j  = xm0 + m + (px + 1 - kw) / 2; // exact
    v8us lo = zero8(), hi = zero8();
    if (i >= 0 && i < HH && j >= 0 && j < WH) {
      const u16* p = rf + ((((size_t)b * HH + i) * WH + j) << 5) + khalf * 8;
      lo = *(const v8us*)(p);
      hi = *(const v8us*)(p + 16);
    }
    v16bf A = mk_frag(lo, hi);
    const u16* wb = wtb + (((kh * 4 + kw) * 32) + (ntile * 16 + m)) * 32 + khalf * 8;
    v16bf Bm = mk_frag(*(const v8us*)(wb), *(const v8us*)(wb + 16));
    acc = wmma_bf16(A, Bm, acc);
  }

  int co = ntile * 16 + m;
  float bias = bt[co];
#pragma unroll
  for (int g = 0; g < 8; ++g) {
    int mm = g + 8 * khalf;               // D layout: lanes>=16 hold M = g+8
    int x  = px + 2 * (xm0 + mm);
    f0[((((size_t)b * HF + y) * WF + x) << 5) + co] = f2bf(acc[g] + bias);
  }
}

// ---------------- stage 2: conv 7x1, 32 -> 49(pad 64) ----------------
// Block = 256 threads (8 waves) per (b, y, 32-px tile). A strip (7 rows x 32 px x 32 ch
// = 14 KB bf16) staged once in LDS via async-to-LDS; waves (mi, ntile) read fragments
// with ds_load_b128. Row-OOB is wave-uniform -> zero fragments (stale LDS never read).
__global__ void k_conv7x1(const u16* __restrict__ f0, const u16* __restrict__ w1b,
                          const float* __restrict__ b1, u16* __restrict__ f1) {
  __shared__ u16 lds0[7 * 32 * 32];       // [r][p][c] bf16

  int id = blockIdx.x;
  int xt = id & 15;  id >>= 4;            // 32-px tiles
  int y  = id & 511; id >>= 9;
  int b  = id;
  int x0 = xt * 32;

  // stage: 896 16-byte chunks
  for (int c = threadIdx.x; c < 896; c += 256) {
    int r = c >> 7, rem = c & 127, p = rem >> 2, q = rem & 3;
    int yy = y + r - 3;
    if (yy >= 0 && yy < HF) {
      const u16* g = f0 + ((((size_t)b * HF + yy) * WF + (x0 + p)) << 5) + q * 8;
      cp16_g2l(&lds0[(r * 32 + p) * 32 + q * 8], g);
    }
  }
  async_fence();
  __syncthreads();

  int lane = threadIdx.x & 31;
  int wid  = threadIdx.x >> 5;            // 8 waves: mi = wid&1, ntile = wid>>1
  int mi = wid & 1, ntile = wid >> 1;
  int m = lane & 15, khalf = lane >> 4;
  int p = mi * 16 + m;                    // pixel within 32-px tile

  v8f acc;
#pragma unroll
  for (int g = 0; g < 8; ++g) acc[g] = 0.f;

#pragma unroll
  for (int r = 0; r < 7; ++r) {
    int yy = y + r - 3;
    v8us lo = zero8(), hi = zero8();
    if (yy >= 0 && yy < HF) {             // wave-uniform branch
      const u16* lp = &lds0[(r * 32 + p) * 32 + khalf * 8];
      lo = *(const v8us*)(lp);
      hi = *(const v8us*)(lp + 16);
    }
    v16bf A = mk_frag(lo, hi);
    const u16* wb = w1b + ((r * 64) + ntile * 16 + m) * 32 + khalf * 8;
    v16bf Bm = mk_frag(*(const v8us*)(wb), *(const v8us*)(wb + 16));
    acc = wmma_bf16(A, Bm, acc);
  }

  int o = ntile * 16 + m;
  float bias = (o < C49) ? b1[o] : 0.f;
#pragma unroll
  for (int g = 0; g < 8; ++g) {
    int mm = g + 8 * khalf;
    int xx = x0 + mi * 16 + mm;
    f1[((((size_t)b * HF + y) * WF + xx) << 6) + o] = f2bf(acc[g] + bias);
  }
}

// ---------------- stage 3: conv 1x7 (49->49) + softmax + 7x7 flow gather ----------------
// Block = 128 threads (4 waves) per (b, y, 16-px tile). A strip (22 px x 64 ch = 2.8 KB)
// staged once in LDS (async); column-OOB chunks zero-filled via ds_store (disjoint).
__global__ void k_conv1x7_softmax(const u16* __restrict__ f1, const u16* __restrict__ w2b,
                                  const float* __restrict__ b2, const float* __restrict__ flow,
                                  float* __restrict__ out) {
  __shared__ u16   lds1[22 * 64];         // [p][c] bf16, p covers x0-3 .. x0+18
  __shared__ float f2buf[16][C64];
  __shared__ float flds[2][7][24];
  __shared__ float red[3][16][8];

  int id = blockIdx.x;
  int xt = id & 31;  id >>= 5;
  int y  = id & 511; id >>= 9;
  int b  = id;
  int x0 = xt * 16;

  // stage f1 strip: 176 16-byte chunks
  for (int c = threadIdx.x; c < 176; c += 128) {
    int p = c >> 3, q = c & 7;
    int xx = x0 - 3 + p;
    if (xx >= 0 && xx < WF) {
      const u16* g = f1 + ((((size_t)b * HF + y) * WF + xx) << 6) + q * 8;
      cp16_g2l(&lds1[p * 64 + q * 8], g);
    } else {
      *(v8us*)(&lds1[p * 64 + q * 8]) = zero8();
    }
  }

  // cooperative flow halo load: rows y-3..y+3, cols x0-3..x0+18, both comps
  for (int idx = threadIdx.x; idx < 2 * 7 * 22; idx += 128) {
    int comp = idx / 154, rem = idx % 154, ki = rem / 22, col = rem % 22;
    int yy = y + ki - 3, xx = x0 + col - 3;
    float v = 0.f;
    if (yy >= 0 && yy < HF && xx >= 0 && xx < WF)
      v = flow[(((size_t)b * 2 + comp) * HF + yy) * WF + xx];
    flds[comp][ki][col] = v;
  }
  async_fence();
  __syncthreads();

  int lane  = threadIdx.x & 31;
  int ntile = threadIdx.x >> 5;
  int m = lane & 15, khalf = lane >> 4;

  // GEMM: K = 7 taps x 64 ch = 14 WMMA steps, A from LDS
  v8f acc;
#pragma unroll
  for (int g = 0; g < 8; ++g) acc[g] = 0.f;

#pragma unroll
  for (int s = 0; s < 7; ++s) {
#pragma unroll
    for (int ch = 0; ch < 2; ++ch) {
      const u16* lp = &lds1[(m + s) * 64 + ch * 32 + khalf * 8];
      v16bf A = mk_frag(*(const v8us*)(lp), *(const v8us*)(lp + 16));
      const u16* wb = w2b + (((s * 2 + ch) * 64) + ntile * 16 + m) * 32 + khalf * 8;
      v16bf Bm = mk_frag(*(const v8us*)(wb), *(const v8us*)(wb + 16));
      acc = wmma_bf16(A, Bm, acc);
    }
  }

  int o = ntile * 16 + m;
  float bias = (o < C49) ? b2[o] : 0.f;
#pragma unroll
  for (int g = 0; g < 8; ++g) {
    int mm = g + 8 * khalf;
    f2buf[mm][o] = acc[g] + bias;
  }
  __syncthreads();

  // softmax over 49 channels + weighted flow gather; 8 partial-threads per pixel
  int p = threadIdx.x & 15;
  int part = threadIdx.x >> 4;
  float qmax = -3.0e38f;
#pragma unroll
  for (int q = 0; q < 7; ++q) {
    int j = part * 7 + q;
    if (j < C49) {
      float d = f2buf[p][j];
      qmax = fmaxf(qmax, -d * d);
    }
  }
  red[0][p][part] = qmax;
  __syncthreads();
  float gmax = red[0][p][0];
#pragma unroll
  for (int k = 1; k < 8; ++k) gmax = fmaxf(gmax, red[0][p][k]);
  __syncthreads();

  float sw = 0.f, sx = 0.f, sy = 0.f;
#pragma unroll
  for (int q = 0; q < 7; ++q) {
    int j = part * 7 + q;
    if (j < C49) {
      float d = f2buf[p][j];
      float e = __expf(-d * d - gmax);
      int ki = j / 7, kj = j % 7;
      sw += e;
      sx += e * flds[0][ki][p + kj];
      sy += e * flds[1][ki][p + kj];
    }
  }
  red[0][p][part] = sw;
  red[1][p][part] = sx;
  red[2][p][part] = sy;
  __syncthreads();

  if (part == 0) {
    float tw = 0.f, tx = 0.f, ty = 0.f;
#pragma unroll
    for (int k = 0; k < 8; ++k) { tw += red[0][p][k]; tx += red[1][p][k]; ty += red[2][p][k]; }
    float inv = 1.f / tw;
    size_t base = (((size_t)b * 2) * HF + y) * WF + (x0 + p);
    out[base] = tx * inv;
    out[base + (size_t)HF * WF] = ty * inv;
  }
}

// ---------------- launcher ----------------
extern "C" void kernel_launch(void* const* d_in, const int* in_sizes, int n_in,
                              void* d_out, int out_size, void* d_ws, size_t ws_size,
                              hipStream_t stream) {
  (void)in_sizes; (void)n_in; (void)out_size; (void)ws_size;
  const float* reg_feat = (const float*)d_in[0];
  const float* flow     = (const float*)d_in[1];
  const float* wt       = (const float*)d_in[2];
  const float* bt       = (const float*)d_in[3];
  const float* w1       = (const float*)d_in[4];
  const float* b1       = (const float*)d_in[5];
  const float* w2       = (const float*)d_in[6];
  const float* b2       = (const float*)d_in[7];
  float* out = (float*)d_out;

  char* ws = (char*)d_ws;
  u16* rf  = (u16*)(ws + OFF_RF);
  u16* f0  = (u16*)(ws + OFF_F0);
  u16* f1  = (u16*)(ws + OFF_F1);
  u16* wtb = (u16*)(ws + OFF_WTB);
  u16* w1b = (u16*)(ws + OFF_W1B);
  u16* w2b = (u16*)(ws + OFF_W2B);

  k_prep_rf<<<(B_N * CH32 * HH * WH) / 256, 256, 0, stream>>>(reg_feat, rf);
  k_prep_wt<<<16384 / 256, 256, 0, stream>>>(wt, wtb);
  k_prep_w1<<<14336 / 256, 256, 0, stream>>>(w1, w1b);
  k_prep_w2<<<28672 / 256, 256, 0, stream>>>(w2, w2b);

  // stage 1: 4*512*2*16*2 = 131072 waves, 8 waves per 256-thread block
  k_convT<<<131072 / 8, 256, 0, stream>>>(rf, wtb, bt, f0);

  // stage 2: 4*512*16 = 32768 blocks of 256 threads
  k_conv7x1<<<32768, 256, 0, stream>>>(f0, w1b, b1, f1);

  // stage 3: 4*512*32 = 65536 blocks of 128 threads
  k_conv1x7_softmax<<<65536, 128, 0, stream>>>(f1, w2b, b2, flow, out);
}
